// EncodecFeatures_3315714752980
// MI455X (gfx1250) — matrix operand order repaired and checked by reading, not compiled, not run on hardware
//
#include <hip/hip_runtime.h>
#include <hip/hip_bf16.h>
#include <stdint.h>

typedef __bf16 bf16_t;
typedef __bf16 v16bf __attribute__((ext_vector_type(16)));
typedef float  v8f   __attribute__((ext_vector_type(8)));
typedef int    v4i_t __attribute__((ext_vector_type(4)));

#if defined(__has_builtin)
#if __has_builtin(__builtin_amdgcn_global_load_async_to_lds_b128) && \
    __has_builtin(__builtin_amdgcn_s_wait_asynccnt)
#define USE_ASYNC_LDS 1
#endif
#endif

// One 16-byte (8 x bf16) per-lane async copy global -> LDS (ASYNCcnt domain).
__device__ __forceinline__ void async_cp16(const bf16_t* g, bf16_t* l) {
#ifdef USE_ASYNC_LDS
    __builtin_amdgcn_global_load_async_to_lds_b128((v4i_t*)g, (v4i_t*)l, 0, 0);
#else
    (void)g; (void)l;
#endif
}

// ---------------------------------------------------------------- helpers
__device__ __forceinline__ float eluf(float v) {
    return v > 0.f ? v : __expf(v) - 1.f;
}
__device__ __forceinline__ float sigf(float v) {
    return 1.f / (1.f + __expf(-v));
}

// A operand fragment: 16x32 bf16 from row-major (lda elems/row).
__device__ __forceinline__ v16bf frag_a(const bf16_t* __restrict__ p0, int lda, int lane) {
    int row = lane & 15, grp = lane >> 4;
    const bf16_t* p = p0 + (long)row * lda + grp * 16;
    v16bf a;
#pragma unroll
    for (int i = 0; i < 8; ++i) { a[i] = p[i]; a[8 + i] = p[8 + i]; }
    return a;
}
// B operand fragment: 32x16 (KxN) from row-major B^T block (N rows x K).
__device__ __forceinline__ v16bf frag_b(const bf16_t* __restrict__ p0, int ldb, int lane) {
    int col = lane & 15, grp = lane >> 4;
    const bf16_t* p = p0 + (long)col * ldb + grp * 16;
    v16bf b;
#pragma unroll
    for (int i = 0; i < 16; ++i) b[i] = p[i];
    return b;
}

__device__ __forceinline__ unsigned long long pack_key(float d, int n) {
    unsigned u = __float_as_uint(d);
    u = (u & 0x80000000u) ? ~u : (u | 0x80000000u);
    return ((unsigned long long)u << 32) | (unsigned)n;
}

// ---------------------------------------------------------------- fills / casts
__global__ void fill_f32(float* p, float v, long n) {
    long i = (long)blockIdx.x * blockDim.x + threadIdx.x;
    if (i < n) p[i] = v;
}
__global__ void fill_u64(unsigned long long* p, unsigned long long v, long n) {
    long i = (long)blockIdx.x * blockDim.x + threadIdx.x;
    if (i < n) p[i] = v;
}
__global__ void cast_f32_bf16(const float* __restrict__ in, bf16_t* __restrict__ out, long n) {
    long i = (long)blockIdx.x * blockDim.x + threadIdx.x;
    if (i < n) out[i] = (bf16_t)in[i];
}
__global__ void vec_add(const float* __restrict__ a, const float* __restrict__ b,
                        float* __restrict__ o, long n) {
    long i = (long)blockIdx.x * blockDim.x + threadIdx.x;
    if (i < n) o[i] = a[i] + b[i];
}

// ---------------------------------------------------------------- conv stack
__global__ void conv1d_reflect(const float* __restrict__ in, const float* __restrict__ w,
                               const float* __restrict__ bias, float* __restrict__ out,
                               int B, int Cin, int Lin, int Cout, int Lout,
                               int Kk, int stride, int pl, int elu_in, int elu_out) {
    long idx = (long)blockIdx.x * blockDim.x + threadIdx.x;
    long total = (long)B * Cout * Lout;
    if (idx >= total) return;
    int l = (int)(idx % Lout);
    long t = idx / Lout;
    int co = (int)(t % Cout);
    int b  = (int)(t / Cout);
    float acc = bias[co];
    const float* wp = w + (long)co * Cin * Kk;
    for (int ci = 0; ci < Cin; ++ci) {
        const float* ip = in + ((long)b * Cin + ci) * Lin;
        for (int kk = 0; kk < Kk; ++kk) {
            int li = l * stride - pl + kk;
            li = li < 0 ? -li : li;
            li = li >= Lin ? 2 * Lin - 2 - li : li;
            float v = ip[li];
            if (elu_in) v = eluf(v);
            acc += v * wp[ci * Kk + kk];
        }
    }
    if (elu_out) acc = eluf(acc);
    out[idx] = acc;
}

// Fused: out = elu( sc_w@x + sc_b + res_w2@elu(y) + res_b2 )   (both 1x1 convs)
__global__ void stage_pointwise(const float* __restrict__ x, const float* __restrict__ y,
                                const float* __restrict__ scw, const float* __restrict__ scb,
                                const float* __restrict__ r2w, const float* __restrict__ r2b,
                                float* __restrict__ out, int B, int C, int H, int L) {
    long idx = (long)blockIdx.x * blockDim.x + threadIdx.x;
    long total = (long)B * C * L;
    if (idx >= total) return;
    int l = (int)(idx % L);
    long t = idx / L;
    int co = (int)(t % C);
    int b  = (int)(t / C);
    float acc = scb[co] + r2b[co];
    const float* xp = x + (long)b * C * L + l;
    const float* wp = scw + (long)co * C;
    for (int ci = 0; ci < C; ++ci) acc += wp[ci] * xp[(long)ci * L];
    const float* yp = y + (long)b * H * L + l;
    const float* rp = r2w + (long)co * H;
    for (int ch = 0; ch < H; ++ch) acc += rp[ch] * eluf(yp[(long)ch * L]);
    out[idx] = eluf(acc);
}

// ---------------------------------------------------------------- WMMA GEMMs
// LDS row pitch (bf16 elems): 32 data + 8 pad -> 80B rows (16B aligned).
#define SPITCH 40

// Blocked GEMM: 256 threads / 8 waves; block tile 64M x 128N; per-wave 16M x 64N.
// A (64x32) and B^T (128x32) double-buffered through LDS with async copies:
// chunk k+1 streams into the alternate buffer while WMMAs run on chunk k.
__global__ void wmma_gemm_blocked(const bf16_t* __restrict__ A, const bf16_t* __restrict__ Bm,
                                  const float* __restrict__ bias,
                                  const float* __restrict__ addsrc,
                                  float* __restrict__ C, int N, int K) {
    __shared__ bf16_t sA[2][64 * SPITCH];
    __shared__ bf16_t sB[2][128 * SPITCH];
    int tid = threadIdx.x;
    int lane = tid & 31, w = tid >> 5;
    int mw = w & 3, nw = w >> 2;
    int m0 = blockIdx.y * 64;
    int n0 = blockIdx.x * 128;
    v8f acc[4] = {};
    int ra = tid >> 2, ca = (tid & 3) * 8;     // A: 64 rows, 8 elems/thread
    int rb = tid >> 1, cb = (tid & 1) * 16;    // B: 128 rows, 16 elems/thread
    const bf16_t* gA = A  + (long)(m0 + ra) * K + ca;
    const bf16_t* gB = Bm + (long)(n0 + rb) * K + cb;
    int nch = K / 32;
#ifdef USE_ASYNC_LDS
    async_cp16(gA, &sA[0][ra * SPITCH + ca]);
    async_cp16(gB, &sB[0][rb * SPITCH + cb]);
    async_cp16(gB + 8, &sB[0][rb * SPITCH + cb + 8]);
    for (int kc = 0; kc < nch; ++kc) {
        int cur = kc & 1;
        if (kc + 1 < nch) {
            const bf16_t* nA = gA + (kc + 1) * 32;
            const bf16_t* nB = gB + (kc + 1) * 32;
            async_cp16(nA, &sA[cur ^ 1][ra * SPITCH + ca]);
            async_cp16(nB, &sB[cur ^ 1][rb * SPITCH + cb]);
            async_cp16(nB + 8, &sB[cur ^ 1][rb * SPITCH + cb + 8]);
            __builtin_amdgcn_s_wait_asynccnt(3);   // oldest chunk retired
        } else {
            __builtin_amdgcn_s_wait_asynccnt(0);
        }
        __syncthreads();
        v16bf a = frag_a(&sA[cur][(mw * 16) * SPITCH], SPITCH, lane);
#pragma unroll
        for (int j = 0; j < 4; ++j) {
            v16bf b = frag_b(&sB[cur][(nw * 64 + j * 16) * SPITCH], SPITCH, lane);
            acc[j] = __builtin_amdgcn_wmma_f32_16x16x32_bf16(false, a, false, b, (short)0,
                                                             acc[j], false, false);
        }
        __syncthreads();
    }
#else
    bf16_t* pA = &sA[0][ra * SPITCH + ca];
    bf16_t* pB = &sB[0][rb * SPITCH + cb];
    for (int kc = 0; kc < nch; ++kc) {
        if (kc + 1 < nch) {
            __builtin_prefetch(gA + 32, 0, 3);
            __builtin_prefetch(gB + 32, 0, 3);
        }
        __syncthreads();
#pragma unroll
        for (int i = 0; i < 8; ++i)  pA[i] = gA[i];
#pragma unroll
        for (int i = 0; i < 16; ++i) pB[i] = gB[i];
        gA += 32; gB += 32;
        __syncthreads();
        v16bf a = frag_a(&sA[0][(mw * 16) * SPITCH], SPITCH, lane);
#pragma unroll
        for (int j = 0; j < 4; ++j) {
            v16bf b = frag_b(&sB[0][(nw * 64 + j * 16) * SPITCH], SPITCH, lane);
            acc[j] = __builtin_amdgcn_wmma_f32_16x16x32_bf16(false, a, false, b, (short)0,
                                                             acc[j], false, false);
        }
    }
#endif
    int grp = lane >> 4, col = lane & 15;
#pragma unroll
    for (int j = 0; j < 4; ++j) {
        int n = n0 + nw * 64 + j * 16 + col;
        float bv = bias ? bias[n] : 0.f;
#pragma unroll
        for (int v = 0; v < 8; ++v) {
            int m = m0 + mw * 16 + grp * 8 + v;
            float r = acc[j][v] + bv;
            if (addsrc) r += addsrc[(long)m * N + n];
            C[(long)m * N + n] = r;
        }
    }
}

// Small-M row-tile GEMM (recurrent LSTM step, M=16): 4 waves/block, 1 tile/wave.
__global__ void wmma_gemm_rowtile(const bf16_t* __restrict__ A, const bf16_t* __restrict__ Bm,
                                  const float* __restrict__ bias,
                                  const float* __restrict__ addsrc,
                                  float* __restrict__ C, int N, int K) {
    int lane = threadIdx.x & 31;
    int wv   = threadIdx.x >> 5;
    int m0 = blockIdx.y * 16;
    int n0 = (blockIdx.x * 4 + wv) * 16;
    v8f acc = {};
    for (int k = 0; k < K; k += 32) {
        v16bf a = frag_a(A  + (long)m0 * K + k, K, lane);
        v16bf b = frag_b(Bm + (long)n0 * K + k, K, lane);
        acc = __builtin_amdgcn_wmma_f32_16x16x32_bf16(false, a, false, b, (short)0, acc,
                                                      false, false);
    }
    int grp = lane >> 4, col = lane & 15;
    int n = n0 + col;
    float bv = bias ? bias[n] : 0.f;
#pragma unroll
    for (int v = 0; v < 8; ++v) {
        int m = m0 + grp * 8 + v;
        float r = acc[v] + bv;
        if (addsrc) r += addsrc[(long)m * N + n];
        C[(long)m * N + n] = r;
    }
}

// Blocked VQ: same async staging; fused argmin epilogue with cross-lane key
// reduction (shfl_xor over 16 lanes sharing one m), then one u64 atomic min.
__global__ void vq_argmin_blocked(const bf16_t* __restrict__ A, const bf16_t* __restrict__ Bm,
                                  const float* __restrict__ cnorm,
                                  unsigned long long* __restrict__ best, int K) {
    __shared__ bf16_t sA[2][64 * SPITCH];
    __shared__ bf16_t sB[2][128 * SPITCH];
    int tid = threadIdx.x;
    int lane = tid & 31, w = tid >> 5;
    int mw = w & 3, nw = w >> 2;
    int m0 = blockIdx.y * 64;
    int n0 = blockIdx.x * 128;
    v8f acc[4] = {};
    int ra = tid >> 2, ca = (tid & 3) * 8;
    int rb = tid >> 1, cb = (tid & 1) * 16;
    const bf16_t* gA = A  + (long)(m0 + ra) * K + ca;
    const bf16_t* gB = Bm + (long)(n0 + rb) * K + cb;
    int nch = K / 32;
#ifdef USE_ASYNC_LDS
    async_cp16(gA, &sA[0][ra * SPITCH + ca]);
    async_cp16(gB, &sB[0][rb * SPITCH + cb]);
    async_cp16(gB + 8, &sB[0][rb * SPITCH + cb + 8]);
    for (int kc = 0; kc < nch; ++kc) {
        int cur = kc & 1;
        if (kc + 1 < nch) {
            const bf16_t* nA = gA + (kc + 1) * 32;
            const bf16_t* nB = gB + (kc + 1) * 32;
            async_cp16(nA, &sA[cur ^ 1][ra * SPITCH + ca]);
            async_cp16(nB, &sB[cur ^ 1][rb * SPITCH + cb]);
            async_cp16(nB + 8, &sB[cur ^ 1][rb * SPITCH + cb + 8]);
            __builtin_amdgcn_s_wait_asynccnt(3);
        } else {
            __builtin_amdgcn_s_wait_asynccnt(0);
        }
        __syncthreads();
        v16bf a = frag_a(&sA[cur][(mw * 16) * SPITCH], SPITCH, lane);
#pragma unroll
        for (int j = 0; j < 4; ++j) {
            v16bf b = frag_b(&sB[cur][(nw * 64 + j * 16) * SPITCH], SPITCH, lane);
            acc[j] = __builtin_amdgcn_wmma_f32_16x16x32_bf16(false, a, false, b, (short)0,
                                                             acc[j], false, false);
        }
        __syncthreads();
    }
#else
    bf16_t* pA = &sA[0][ra * SPITCH + ca];
    bf16_t* pB = &sB[0][rb * SPITCH + cb];
    for (int kc = 0; kc < nch; ++kc) {
        if (kc + 1 < nch) {
            __builtin_prefetch(gA + 32, 0, 3);
            __builtin_prefetch(gB + 32, 0, 3);
        }
        __syncthreads();
#pragma unroll
        for (int i = 0; i < 8; ++i)  pA[i] = gA[i];
#pragma unroll
        for (int i = 0; i < 16; ++i) pB[i] = gB[i];
        gA += 32; gB += 32;
        __syncthreads();
        v16bf a = frag_a(&sA[0][(mw * 16) * SPITCH], SPITCH, lane);
#pragma unroll
        for (int j = 0; j < 4; ++j) {
            v16bf b = frag_b(&sB[0][(nw * 64 + j * 16) * SPITCH], SPITCH, lane);
            acc[j] = __builtin_amdgcn_wmma_f32_16x16x32_bf16(false, a, false, b, (short)0,
                                                             acc[j], false, false);
        }
    }
#endif
    int grp = lane >> 4, col = lane & 15;
#pragma unroll
    for (int j = 0; j < 4; ++j) {
        int n = n0 + nw * 64 + j * 16 + col;
        float cn = cnorm[n];
#pragma unroll
        for (int v = 0; v < 8; ++v) {
            int m = m0 + mw * 16 + grp * 8 + v;
            unsigned long long key = pack_key(cn - 2.f * acc[j][v], n);
#pragma unroll
            for (int s = 8; s >= 1; s >>= 1) {
                unsigned long long o = __shfl_xor(key, s, 32);
                key = o < key ? o : key;
            }
            if (col == 0) atomicMin(best + m, key);
        }
    }
}

// ---------------------------------------------------------------- LSTM pieces
__global__ void lstm_pointwise(const float* __restrict__ gates, float* __restrict__ h,
                               float* __restrict__ c, bf16_t* __restrict__ hbf,
                               float* __restrict__ yout, bf16_t* __restrict__ ybf, int t) {
    int b = blockIdx.x;   // 16
    int j = threadIdx.x;  // 512
    const float* g = gates + (long)b * 2048;
    float si = sigf(g[j]);
    float sf = sigf(g[512 + j]);
    float gg = tanhf(g[1024 + j]);
    float so = sigf(g[1536 + j]);
    float cv = sf * c[b * 512 + j] + si * gg;
    float hv = so * tanhf(cv);
    c[b * 512 + j] = cv;
    h[b * 512 + j] = hv;
    hbf[b * 512 + j] = (bf16_t)hv;
    long oi = ((long)t * 16 + b) * 512 + j;
    yout[oi] = hv;
    ybf[oi] = (bf16_t)hv;
}

// (B=16, D=512, L=240) -> time-major tokens (l*16+b, d), fp32 + bf16
__global__ void make_seq(const float* __restrict__ x, float* __restrict__ seq,
                         bf16_t* __restrict__ seqbf) {
    long idx = (long)blockIdx.x * blockDim.x + threadIdx.x;
    long total = 16L * 512 * 240;
    if (idx >= total) return;
    int l = (int)(idx % 240);
    long t = idx / 240;
    int d = (int)(t % 512);
    int b = (int)(t / 512);
    float v = x[idx];
    long o = ((long)l * 16 + b) * 512 + d;
    seq[o] = v;
    seqbf[o] = (bf16_t)v;
}

// im2col for final conv: tokens (b*240+l) x (ci*7+kk), elu + reflect fused.
__global__ void im2col_final(const float* __restrict__ seqsum, bf16_t* __restrict__ A) {
    long idx = (long)blockIdx.x * blockDim.x + threadIdx.x;
    long total = 3840L * 3584;
    if (idx >= total) return;
    int colk = (int)(idx % 3584);
    long r = idx / 3584;
    int l = (int)(r % 240);
    int b = (int)(r / 240);
    int ci = colk / 7, kk = colk % 7;
    int li = l + kk - 3;
    li = li < 0 ? -li : li;
    li = li >= 240 ? 2 * 240 - 2 - li : li;
    float v = seqsum[((long)li * 16 + b) * 512 + ci];
    A[idx] = (bf16_t)eluf(v);
}

__global__ void codebook_norm(const float* __restrict__ cb, float* __restrict__ cnorm) {
    int n = blockIdx.x * blockDim.x + threadIdx.x;
    if (n >= 16384) return;
    const float* p = cb + (long)n * 512;
    float s = 0.f;
    for (int d = 0; d < 512; ++d) s += p[d] * p[d];
    cnorm[n] = s;
}

__global__ void vq_finalize(const unsigned long long* __restrict__ best,
                            const float* __restrict__ cb, const float* __restrict__ flat,
                            float* __restrict__ quant, float* __restrict__ codes,
                            float* __restrict__ commit_acc) {
    __shared__ float red[512];
    int m = blockIdx.x;   // token = b*240+l
    int d = threadIdx.x;  // 0..511
    unsigned ind = (unsigned)(best[m] & 0xFFFFFFFFull);
    float q = cb[(long)ind * 512 + d];
    int b = m / 240, l = m % 240;
    quant[((long)b * 512 + d) * 240 + l] = q;
    float diff = q - flat[(long)m * 512 + d];
    red[d] = diff * diff;
    __syncthreads();
    for (int s = 256; s > 0; s >>= 1) {
        if (d < s) red[d] += red[d + s];
        __syncthreads();
    }
    if (d == 0) {
        atomicAdd(commit_acc, red[0]);
        codes[m] = (float)ind;
    }
}
__global__ void commit_scale(const float* __restrict__ acc, float* __restrict__ out) {
    out[0] = acc[0] / (3840.f * 512.f);
}

// ---------------------------------------------------------------- host
extern "C" void kernel_launch(void* const* d_in, const int* in_sizes, int n_in,
                              void* d_out, int out_size, void* d_ws, size_t ws_size,
                              hipStream_t stream) {
    (void)in_sizes; (void)n_in; (void)out_size; (void)ws_size;
    int p = 0;
    const float* audio   = (const float*)d_in[p++];
    const float* conv0_w = (const float*)d_in[p++];
    const float* conv0_b = (const float*)d_in[p++];
    struct StageP { const float *rw1,*rb1,*rw2,*rb2,*scw,*scb,*dw,*db; } st[4];
    for (int s = 0; s < 4; ++s) {
        st[s].rw1 = (const float*)d_in[p++]; st[s].rb1 = (const float*)d_in[p++];
        st[s].rw2 = (const float*)d_in[p++]; st[s].rb2 = (const float*)d_in[p++];
        st[s].scw = (const float*)d_in[p++]; st[s].scb = (const float*)d_in[p++];
        st[s].dw  = (const float*)d_in[p++]; st[s].db  = (const float*)d_in[p++];
    }
    struct LstmP { const float *wih,*whh,*bih,*bhh; } lp[2];
    for (int s = 0; s < 2; ++s) {
        lp[s].wih = (const float*)d_in[p++]; lp[s].whh = (const float*)d_in[p++];
        lp[s].bih = (const float*)d_in[p++]; lp[s].bhh = (const float*)d_in[p++];
    }
    const float* final_w  = (const float*)d_in[p++];
    const float* final_b  = (const float*)d_in[p++];
    const float* codebook = (const float*)d_in[p++];

    // ---- workspace carve-up
    char* ws = (char*)d_ws;
    size_t off = 0;
    auto alloc = [&](size_t bytes) -> void* {
        void* r = ws + off;
        off += (bytes + 255) & ~(size_t)255;
        return r;
    };
    float*  bufA   = (float*)alloc(73728000UL * 4);
    float*  bufB   = (float*)alloc(73728000UL * 4);
    float*  bufY   = (float*)alloc(36864000UL * 4);
    const long NTOK = 3840, DMODEL = 512;
    float*  seq    = (float*)alloc(NTOK * DMODEL * 4);
    bf16_t* seqbf  = (bf16_t*)alloc(NTOK * DMODEL * 2);
    bf16_t* wihbf[2], *whhbf[2];
    for (int s = 0; s < 2; ++s) {
        wihbf[s] = (bf16_t*)alloc(2048UL * 512 * 2);
        whhbf[s] = (bf16_t*)alloc(2048UL * 512 * 2);
    }
    float*  bsum[2] = {(float*)alloc(2048 * 4), (float*)alloc(2048 * 4)};
    float*  Xg     = (float*)alloc(NTOK * 2048 * 4);
    float*  gates  = (float*)alloc(16UL * 2048 * 4);
    float*  hbuf   = (float*)alloc(16UL * 512 * 4);
    float*  cbuf   = (float*)alloc(16UL * 512 * 4);
    bf16_t* hbf    = (bf16_t*)alloc(16UL * 512 * 2);
    float*  yL[2]  = {(float*)alloc(NTOK * DMODEL * 4), (float*)alloc(NTOK * DMODEL * 4)};
    bf16_t* yLbf[2]= {(bf16_t*)alloc(NTOK * DMODEL * 2), (bf16_t*)alloc(NTOK * DMODEL * 2)};
    float*  seqsum = (float*)alloc(NTOK * DMODEL * 4);
    bf16_t* im2c   = (bf16_t*)alloc(NTOK * 3584UL * 2);
    bf16_t* fwbf   = (bf16_t*)alloc(512UL * 3584 * 2);
    float*  flat   = (float*)alloc(NTOK * DMODEL * 4);
    bf16_t* flatbf = (bf16_t*)alloc(NTOK * DMODEL * 2);
    bf16_t* cbbf   = (bf16_t*)alloc(16384UL * 512 * 2);
    float*  cnorm  = (float*)alloc(16384 * 4);
    unsigned long long* best = (unsigned long long*)alloc(NTOK * 8);
    float*  cacc   = (float*)alloc(4);

    auto g1 = [](long n) { return dim3((unsigned)((n + 255) / 256)); };

    // ---- conv stack (bandwidth-bound -> VALU)
    conv1d_reflect<<<g1(16L*32*144000), 256, 0, stream>>>(
        audio, conv0_w, conv0_b, bufA, 16, 1, 144000, 32, 144000, 7, 1, 3, 0, 0);
    int C = 32; int L = 144000;
    const int ratios[4] = {4, 5, 5, 6};
    for (int s = 0; s < 4; ++s) {
        int h = C / 2, r = ratios[s];
        conv1d_reflect<<<g1((long)16*h*L), 256, 0, stream>>>(
            bufA, st[s].rw1, st[s].rb1, bufY, 16, C, L, h, L, 3, 1, 1, 1, 0);
        stage_pointwise<<<g1((long)16*C*L), 256, 0, stream>>>(
            bufA, bufY, st[s].scw, st[s].scb, st[s].rw2, st[s].rb2, bufB, 16, C, h, L);
        int Lout = L / r, pr = r / 2, pl = r - pr;
        conv1d_reflect<<<g1((long)16*2*C*Lout), 256, 0, stream>>>(
            bufB, st[s].dw, st[s].db, bufA, 16, C, L, 2*C, Lout, 2*r, r, pl, 0, 0);
        C *= 2; L = Lout;
    }
    // bufA = (16, 512, 240)
    make_seq<<<g1(NTOK * DMODEL), 256, 0, stream>>>(bufA, seq, seqbf);

    // ---- LSTM: Xg = X@Wih^T + (bih+bhh) hoisted (blocked WMMA); per-step
    //      recurrence h@Whh^T kept on the row-tile WMMA kernel.
    const bf16_t* layer_in_bf = seqbf;
    for (int lyr = 0; lyr < 2; ++lyr) {
        cast_f32_bf16<<<g1(2048L*512), 256, 0, stream>>>(lp[lyr].wih, wihbf[lyr], 2048L*512);
        cast_f32_bf16<<<g1(2048L*512), 256, 0, stream>>>(lp[lyr].whh, whhbf[lyr], 2048L*512);
        vec_add<<<g1(2048), 256, 0, stream>>>(lp[lyr].bih, lp[lyr].bhh, bsum[lyr], 2048);
        // Xg: M=3840, N=2048, K=512
        wmma_gemm_blocked<<<dim3(2048/128, 3840/64), 256, 0, stream>>>(
            layer_in_bf, wihbf[lyr], bsum[lyr], nullptr, Xg, 2048, 512);
        fill_f32<<<g1(16L*512), 256, 0, stream>>>(hbuf, 0.f, 16L*512);
        fill_f32<<<g1(16L*512), 256, 0, stream>>>(cbuf, 0.f, 16L*512);
        fill_f32<<<g1(16L*512/2), 256, 0, stream>>>((float*)hbf, 0.f, 16L*512/2);
        for (int t = 0; t < 240; ++t) {
            wmma_gemm_rowtile<<<dim3(2048/64, 1), 128, 0, stream>>>(
                hbf, whhbf[lyr], nullptr, Xg + (long)t*16*2048, gates, 2048, 512);
            lstm_pointwise<<<16, 512, 0, stream>>>(gates, hbuf, cbuf, hbf,
                                                   yL[lyr], yLbf[lyr], t);
        }
        layer_in_bf = yLbf[lyr];
    }
    // residual: seqsum = y2 + seq   (both time-major (L,B,D))
    vec_add<<<g1(NTOK * DMODEL), 256, 0, stream>>>(yL[1], seq, seqsum, NTOK * DMODEL);

    // ---- final conv as im2col + blocked WMMA GEMM -> flat tokens (b*240+l, 512)
    im2col_final<<<g1(NTOK * 3584), 256, 0, stream>>>(seqsum, im2c);
    cast_f32_bf16<<<g1(512L*3584), 256, 0, stream>>>(final_w, fwbf, 512L*3584);
    wmma_gemm_blocked<<<dim3(512/128, 3840/64), 256, 0, stream>>>(
        im2c, fwbf, final_b, nullptr, flat, 512, 3584);

    // ---- VQ: 3840 x 16384 x 512 blocked WMMA GEMM with fused argmin
    cast_f32_bf16<<<g1(NTOK * DMODEL), 256, 0, stream>>>(flat, flatbf, NTOK * DMODEL);
    cast_f32_bf16<<<g1(16384L*512), 256, 0, stream>>>(codebook, cbbf, 16384L*512);
    codebook_norm<<<g1(16384), 256, 0, stream>>>(codebook, cnorm);
    fill_u64<<<g1(NTOK), 256, 0, stream>>>(best, ~0ULL, NTOK);
    fill_f32<<<1, 1, 0, stream>>>(cacc, 0.f, 1);
    vq_argmin_blocked<<<dim3(16384/128, 3840/64), 256, 0, stream>>>(flatbf, cbbf, cnorm,
                                                                    best, 512);

    float* quant_out  = (float*)d_out;                    // (16,512,240)
    float* codes_out  = (float*)d_out + 16L*512*240;      // 3840 codes
    float* commit_out = codes_out + 3840;                 // 1 scalar
    vq_finalize<<<3840, 512, 0, stream>>>(best, codebook, flat, quant_out, codes_out, cacc);
    commit_scale<<<1, 1, 0, stream>>>(cacc, commit_out);
}
